// conv_approximation_35837207118638
// MI455X (gfx1250) — matrix operand order repaired and checked by reading, not compiled, not run on hardware
//
#include <hip/hip_runtime.h>
#include <hip/hip_bf16.h>

typedef __attribute__((ext_vector_type(16))) __bf16        v16bf;
typedef __attribute__((ext_vector_type(16))) unsigned short v16u;
typedef __attribute__((ext_vector_type(8)))  float          v8f;

#define BBATCH 4
#define NSEQ   4096
#define HHEADS 8
#define DDIM   64
#define MFEAT  256
#define RROWS  (BBATCH * NSEQ * HHEADS)   // 131072
#define KSPLIT 16

__device__ __forceinline__ unsigned short f2bf(float f) {
    unsigned int u = __float_as_uint(f);
    u += 0x7FFFu + ((u >> 16) & 1u);      // round-to-nearest-even
    return (unsigned short)(u >> 16);
}

__device__ __forceinline__ v8f bf_wmma(v16u a, v16u b, v8f c) {
    return __builtin_amdgcn_wmma_f32_16x16x32_bf16(
        /*neg_a=*/false, __builtin_bit_cast(v16bf, a),
        /*neg_b=*/false, __builtin_bit_cast(v16bf, b),
        /*c_mod=*/(short)0, c, /*reuse_a=*/false, /*reuse_b=*/false);
}

// ---------------------------------------------------------------------------
// 1) scale input, emit bf16 copy + per-row diag = 0.5 * sum(data^2)
// ---------------------------------------------------------------------------
__global__ __launch_bounds__(256) void prep_kernel(const float* __restrict__ src,
                                                   unsigned short* __restrict__ abf,
                                                   float* __restrict__ diag) {
    __shared__ float wsum[8];
    int t = threadIdx.x;
    int rbase = blockIdx.x * 4;
    int rl = t >> 6, c = t & 63;
    int r = rbase + rl;
    float val = src[r * 64 + c] * 0.35355339059327373f;   // 64^-0.25
    abf[r * 64 + c] = f2bf(val);
    float s = val * val;
    #pragma unroll
    for (int m = 16; m >= 1; m >>= 1) s += __shfl_xor(s, m, 32);
    if ((t & 31) == 0) wsum[t >> 5] = s;
    __syncthreads();
    if (t < 4) diag[rbase + t] = 0.5f * (wsum[2 * t] + wsum[2 * t + 1]);
}

__global__ __launch_bounds__(256) void convp_kernel(const float* __restrict__ p,
                                                    unsigned short* __restrict__ pbf) {
    int i = blockIdx.x * 256 + threadIdx.x;
    if (i < MFEAT * DDIM) pbf[i] = f2bf(p[i]);
}

// ---------------------------------------------------------------------------
// 2) projection GEMM: dd[r, m] = A[r,:64] . P[m,:64]   (bf16 WMMA, K=64)
//    also per-row max over m.
// ---------------------------------------------------------------------------
__global__ __launch_bounds__(256) void gemm_proj_kernel(
        const unsigned short* __restrict__ abf, const unsigned short* __restrict__ pbf,
        float* __restrict__ dd, float* __restrict__ rowmax) {
    __shared__ unsigned short lA[16 * 64];     // 2 KB
    __shared__ unsigned short lP[256 * 64];    // 32 KB
    __shared__ float          ldd[16 * 264];   // ~16.9 KB (padded)
    int t = threadIdx.x;
    int rbase = blockIdx.x * 16;

    // prefetch output row region we will stream later (write-allocate hint)
    __builtin_prefetch(&dd[(size_t)rbase * 256 + t * 4], 1, 1);

    #pragma unroll
    for (int j = 0; j < 4; ++j) {
        int idx = t + j * 256;
        lA[idx] = abf[rbase * 64 + idx];
    }
    for (int j = 0; j < 64; ++j) lP[t + j * 256] = pbf[t + j * 256];
    __syncthreads();

    int lane = t & 31, wid = t >> 5;
    int aM = lane & 15, aKb = (lane >> 4) * 8;
    v16u a0, a1;
    #pragma unroll
    for (int i = 0; i < 16; ++i) {
        int k = aKb * 2 + i;                         // contiguous k = 16h + i
        a0[i] = lA[aM * 64 + k];
        a1[i] = lA[aM * 64 + 32 + k];
    }
    #pragma unroll
    for (int ct2 = 0; ct2 < 2; ++ct2) {
        int ct = wid * 2 + ct2;                      // 8 waves x 2 = 16 col tiles
        int bN = ct * 16 + (lane & 15);
        int bKb = (lane >> 4) * 16;
        v16u b0, b1;
        #pragma unroll
        for (int i = 0; i < 16; ++i) {               // B[k][n] = P[n][k]
            b0[i] = lP[bN * 64 + bKb + i];
            b1[i] = lP[bN * 64 + 32 + bKb + i];
        }
        v8f c = {};
        c = bf_wmma(a0, b0, c);
        c = bf_wmma(a1, b1, c);
        #pragma unroll
        for (int v = 0; v < 8; ++v) {
            int M = v + ((lane >> 4) << 3);
            ldd[M * 264 + ct * 16 + (lane & 15)] = c[v];
        }
    }
    __syncthreads();
    {   // per-row max over m=256
        int row = t >> 4, ch = t & 15;
        float m = -3.0e38f;
        #pragma unroll
        for (int i = 0; i < 16; ++i) m = fmaxf(m, ldd[row * 264 + ch * 16 + i]);
        #pragma unroll
        for (int s = 8; s >= 1; s >>= 1) m = fmaxf(m, __shfl_xor(m, s, 32));
        if (ch == 0) rowmax[rbase + row] = m;
    }
    #pragma unroll
    for (int j = 0; j < 16; ++j) {                  // coalesced writeback
        int idx = t + j * 256;
        int row = idx >> 8, col = idx & 255;
        dd[(size_t)(rbase + row) * 256 + col] = ldd[row * 264 + col];
    }
}

// ---------------------------------------------------------------------------
// 3) key stabilizer: max over n per (b,h)
// ---------------------------------------------------------------------------
__global__ __launch_bounds__(256) void keystab_kernel(const float* __restrict__ rowmax2,
                                                      float* __restrict__ keymax) {
    __shared__ float red[256];
    int t = threadIdx.x;
    int bb = blockIdx.x >> 3, hh = blockIdx.x & 7;
    float m = -3.0e38f;
    for (int nn = t; nn < NSEQ; nn += 256)
        m = fmaxf(m, rowmax2[(bb * NSEQ + nn) * HHEADS + hh]);
    red[t] = m;
    __syncthreads();
    for (int s = 128; s >= 1; s >>= 1) {
        if (t < s) red[t] = fmaxf(red[t], red[t + s]);
        __syncthreads();
    }
    if (t == 0) keymax[blockIdx.x] = red[0];
}

// ---------------------------------------------------------------------------
// 4) exp pass, in place: dd = ratio*(exp(dd - diag - stab) + eps)
// ---------------------------------------------------------------------------
__global__ __launch_bounds__(256) void exp_kernel(float* __restrict__ dd,
                                                  const float* __restrict__ diag,
                                                  const float* __restrict__ stab,
                                                  int perbh) {
    long long i4 = (long long)blockIdx.x * 256 + threadIdx.x;
    long long base = i4 * 4;
    int r = (int)(base >> 8);
    float s;
    if (perbh) {
        int bb = r >> 15;            // r / (NSEQ*HHEADS)
        int hh = r & 7;
        s = stab[bb * 8 + hh];
    } else {
        s = stab[r];
    }
    float dsub = diag[r] + s;
    float4 v = *(const float4*)(dd + base);
    v.x = 0.0625f * (__expf(v.x - dsub) + 1e-6f);
    v.y = 0.0625f * (__expf(v.y - dsub) + 1e-6f);
    v.z = 0.0625f * (__expf(v.z - dsub) + 1e-6f);
    v.w = 0.0625f * (__expf(v.w - dsub) + 1e-6f);
    *(float4*)(dd + base) = v;
}

// ---------------------------------------------------------------------------
// 5) v2x partial = v2p^T . x over a K-slice of n; fused column sums (v2_sum)
//    grid = 32 bh * KSPLIT; deterministic (no global float atomics)
// ---------------------------------------------------------------------------
__global__ __launch_bounds__(256) void gemm_kv_kernel(
        const float* __restrict__ v2p, const float* __restrict__ x,
        float* __restrict__ v2xp, float* __restrict__ v2sp) {
    __shared__ unsigned short lV[32 * 256];   // 16 KB
    __shared__ unsigned short lX[32 * 64];    // 4 KB
    int t = threadIdx.x;
    int blk = blockIdx.x;
    int bh = blk / KSPLIT, ks = blk % KSPLIT;
    int bb = bh >> 3, hh = bh & 7;
    int kbeg = ks * (NSEQ / KSPLIT);
    int kend = kbeg + NSEQ / KSPLIT;
    int lane = t & 31, wid = t >> 5;
    float vsum = 0.0f;
    v8f acc[2][4];
    #pragma unroll
    for (int i = 0; i < 2; ++i)
        #pragma unroll
        for (int j = 0; j < 4; ++j) acc[i][j] = (v8f){};

    for (int k0 = kbeg; k0 < kend; k0 += 32) {
        __syncthreads();
        #pragma unroll 4
        for (int j = 0; j < 32; ++j) {                      // thread t owns column m=t
            int rr = (bb * NSEQ + k0 + j) * HHEADS + hh;
            float val = v2p[(size_t)rr * 256 + t];
            lV[j * 256 + t] = f2bf(val);
            vsum += val;
        }
        #pragma unroll
        for (int j = 0; j < 8; ++j) {
            int idx = j * 256 + t;
            int np = idx >> 6, cc = idx & 63;
            int rr = (bb * NSEQ + k0 + np) * HHEADS + hh;
            lX[idx] = f2bf(x[(size_t)rr * 64 + cc]);
        }
        // prefetch next chunk (global_prefetch_b8) so HBM latency overlaps WMMA
        if (k0 + 32 < kend) {
            int jn = t >> 3;                                 // 32 rows, 8 lines each
            int rr = (bb * NSEQ + k0 + 32 + jn) * HHEADS + hh;
            __builtin_prefetch(&v2p[(size_t)rr * 256 + (t & 7) * 32], 0, 1);
            if (t < 32) {
                int rx = (bb * NSEQ + k0 + 32 + t) * HHEADS + hh;
                __builtin_prefetch(&x[(size_t)rx * 64], 0, 1);
            }
        }
        __syncthreads();
        v16u bfr[4];
        int bKb = (lane >> 4) * 16;
        #pragma unroll
        for (int dt = 0; dt < 4; ++dt)
            #pragma unroll
            for (int i = 0; i < 16; ++i)
                bfr[dt][i] = lX[(bKb + i) * 64 + dt * 16 + (lane & 15)];
        int aKb = (lane >> 4) * 8;
        #pragma unroll
        for (int mi = 0; mi < 2; ++mi) {
            int mt = wid * 2 + mi;
            v16u a;
            #pragma unroll
            for (int i = 0; i < 16; ++i) {
                int k = aKb * 2 + i;
                a[i] = lV[k * 256 + mt * 16 + (lane & 15)];   // transposed read
            }
            #pragma unroll
            for (int dt = 0; dt < 4; ++dt)
                acc[mi][dt] = bf_wmma(a, bfr[dt], acc[mi][dt]);
        }
    }
    #pragma unroll
    for (int mi = 0; mi < 2; ++mi) {
        int mt = wid * 2 + mi;
        #pragma unroll
        for (int dt = 0; dt < 4; ++dt)
            #pragma unroll
            for (int v = 0; v < 8; ++v) {
                int mm = mt * 16 + v + ((lane >> 4) << 3);
                int cc = dt * 16 + (lane & 15);
                v2xp[(size_t)blk * MFEAT * DDIM + mm * 64 + cc] = acc[mi][dt][v];
            }
    }
    v2sp[blk * 256 + t] = vsum;
}

__global__ __launch_bounds__(256) void reduce_kv_kernel(const float* __restrict__ v2xp,
                                                        float* __restrict__ v2x) {
    int i = blockIdx.x * 256 + threadIdx.x;   // over 32*16384
    int bh = i >> 14, off = i & 16383;
    float s = 0.0f;
    #pragma unroll
    for (int ks = 0; ks < KSPLIT; ++ks)
        s += v2xp[(size_t)(bh * KSPLIT + ks) * 16384 + off];
    v2x[i] = s;
}

__global__ __launch_bounds__(256) void reduce_vs_kernel(const float* __restrict__ v2sp,
                                                        float* __restrict__ v2sum) {
    int i = blockIdx.x * 256 + threadIdx.x;  // over 32*256
    int bh = i >> 8, off = i & 255;
    float s = 0.0f;
    #pragma unroll
    for (int ks = 0; ks < KSPLIT; ++ks)
        s += v2sp[(bh * KSPLIT + ks) * 256 + off];
    v2sum[i] = s;
}

// ---------------------------------------------------------------------------
// 6) out = (v1p . v2x) / (v1p . v2sum), fused divide
// ---------------------------------------------------------------------------
__global__ __launch_bounds__(256) void gemm_out_kernel(
        const float* __restrict__ v1p, const float* __restrict__ v2x,
        const float* __restrict__ v2sum, float* __restrict__ out) {
    __shared__ unsigned short lV1[32 * 256];   // 16 KB
    __shared__ unsigned short lW[256 * 64];    // 32 KB
    __shared__ float lS[256];
    __shared__ float lO2[32];
    int t = threadIdx.x;
    int bh = blockIdx.x >> 7;                  // 128 n-tiles of 32 per (b,h)
    int nt = blockIdx.x & 127;
    int bb = bh >> 3, hh = bh & 7;
    int nbase = nt * 32;

    // prefetch the v1p rows this block will consume (1 KB rows, h*d-strided)
    {
        int jn = t >> 3;
        int rr = (bb * NSEQ + nbase + jn) * HHEADS + hh;
        __builtin_prefetch(&v1p[(size_t)rr * 256 + (t & 7) * 32], 0, 1);
    }

    lS[t] = v2sum[bh * 256 + t];
    if (t < 32) lO2[t] = 0.0f;
    __syncthreads();

    for (int j = 0; j < 64; ++j) {
        int idx = j * 256 + t;
        lW[idx] = f2bf(v2x[(size_t)bh * MFEAT * DDIM + idx]);
    }
    #pragma unroll 4
    for (int j = 0; j < 32; ++j) {
        int rr = (bb * NSEQ + nbase + j) * HHEADS + hh;
        float val = v1p[(size_t)rr * 256 + t];
        lV1[j * 256 + t] = f2bf(val);
        atomicAdd(&lO2[j], val * lS[t]);       // ds_add_f32
    }
    __syncthreads();

    int lane = t & 31, wid = t >> 5;
    int rt = wid >> 2, dt = wid & 3;           // 8 waves -> 2 row-tiles x 4 d-tiles
    v8f acc = {};
    int aM = rt * 16 + (lane & 15);
    int aKb = (lane >> 4) * 8;
    int bKb = (lane >> 4) * 16;
    int bN = dt * 16 + (lane & 15);
    #pragma unroll
    for (int k0 = 0; k0 < 256; k0 += 32) {
        v16u a, bfr;
        #pragma unroll
        for (int i = 0; i < 16; ++i) {
            int k = k0 + aKb * 2 + i;
            a[i] = lV1[aM * 256 + k];
        }
        #pragma unroll
        for (int i = 0; i < 16; ++i)
            bfr[i] = lW[(k0 + bKb + i) * 64 + bN];
        acc = bf_wmma(a, bfr, acc);
    }
    #pragma unroll
    for (int v = 0; v < 8; ++v) {
        int row = rt * 16 + v + ((lane >> 4) << 3);
        int rr = (bb * NSEQ + nbase + row) * HHEADS + hh;
        out[(size_t)rr * 64 + dt * 16 + (lane & 15)] = acc[v] / lO2[row];
    }
}

// ---------------------------------------------------------------------------
extern "C" void kernel_launch(void* const* d_in, const int* in_sizes, int n_in,
                              void* d_out, int out_size, void* d_ws, size_t ws_size,
                              hipStream_t stream) {
    const float* x   = (const float*)d_in[0];
    const float* nv1 = (const float*)d_in[1];
    const float* nv2 = (const float*)d_in[2];
    const float* prj = (const float*)d_in[3];

    float* out = (float*)d_out;                          // [R,64]
    float* v1p = out + (size_t)RROWS * DDIM;             // [R,256]
    float* v2p = v1p + (size_t)RROWS * MFEAT;            // [R,256]

    char* w = (char*)d_ws;
    auto alloc = [&](size_t bytes) {
        char* p = w;
        w += (bytes + 255) & ~(size_t)255;
        return (void*)p;
    };
    unsigned short* a1bf = (unsigned short*)alloc((size_t)RROWS * DDIM * 2);
    unsigned short* a2bf = (unsigned short*)alloc((size_t)RROWS * DDIM * 2);
    unsigned short* pbf  = (unsigned short*)alloc((size_t)MFEAT * DDIM * 2);
    float* diag1 = (float*)alloc((size_t)RROWS * 4);
    float* diag2 = (float*)alloc((size_t)RROWS * 4);
    float* rmax1 = (float*)alloc((size_t)RROWS * 4);
    float* rmax2 = (float*)alloc((size_t)RROWS * 4);
    float* kmax  = (float*)alloc(32 * 4);
    float* v2x   = (float*)alloc((size_t)32 * MFEAT * DDIM * 4);
    float* v2s   = (float*)alloc((size_t)32 * MFEAT * 4);
    float* v2xp  = (float*)alloc((size_t)32 * KSPLIT * MFEAT * DDIM * 4);
    float* v2sp  = (float*)alloc((size_t)32 * KSPLIT * MFEAT * 4);

    prep_kernel<<<RROWS / 4, 256, 0, stream>>>(nv1, a1bf, diag1);
    prep_kernel<<<RROWS / 4, 256, 0, stream>>>(nv2, a2bf, diag2);
    convp_kernel<<<(MFEAT * DDIM + 255) / 256, 256, 0, stream>>>(prj, pbf);

    gemm_proj_kernel<<<RROWS / 16, 256, 0, stream>>>(a1bf, pbf, v1p, rmax1);
    gemm_proj_kernel<<<RROWS / 16, 256, 0, stream>>>(a2bf, pbf, v2p, rmax2);

    keystab_kernel<<<32, 256, 0, stream>>>(rmax2, kmax);

    exp_kernel<<<RROWS / 4, 256, 0, stream>>>(v1p, diag1, rmax1, 0);
    exp_kernel<<<RROWS / 4, 256, 0, stream>>>(v2p, diag2, kmax, 1);

    gemm_kv_kernel<<<32 * KSPLIT, 256, 0, stream>>>(v2p, x, v2xp, v2sp);
    reduce_kv_kernel<<<(32 * MFEAT * DDIM) / 256, 256, 0, stream>>>(v2xp, v2x);
    reduce_vs_kernel<<<(32 * MFEAT) / 256, 256, 0, stream>>>(v2sp, v2s);

    gemm_out_kernel<<<32 * (NSEQ / 32), 256, 0, stream>>>(v1p, v2x, v2s, out);
}
